// VRPActionNet_29231547417133
// MI455X (gfx1250) — matrix operand order repaired and checked by reading, not compiled, not run on hardware
//
#include <hip/hip_runtime.h>

typedef __attribute__((ext_vector_type(16))) _Float16 half16;
typedef __attribute__((ext_vector_type(8)))  _Float16 half8;
typedef __attribute__((ext_vector_type(4)))  _Float16 half4_t;
typedef __attribute__((ext_vector_type(8)))  float    v8f;

#define H_DIM 128
#define B_DIM 64
#define N_DIM 100
#define M_DIM 2000
#define TM    32          // moves per block tile
#define HSTR  (H_DIM + 8) // padded LDS row stride (halves): conflict-free b128 loads

// ---------------------------------------------------------------------------
// Weight prep: f32 [K][C] -> f16 transposed [C][K] so a WMMA B-fragment is two
// contiguous 16B reads per lane (N = lane%16, K runs 8h..8h+7 and 16+8h..16+8h+7).
// ---------------------------------------------------------------------------
__global__ void cvt_transpose_f16(const float* __restrict__ W,
                                  _Float16* __restrict__ Wt, int K, int C) {
    int i = blockIdx.x * blockDim.x + threadIdx.x;
    if (i < K * C) {
        int c = i / K, k = i % K;
        Wt[i] = (_Float16)W[(size_t)k * C + c];
    }
}

// ---------------------------------------------------------------------------
// One GEMM layer on a 32xK LDS tile: Y[32][128] = act(X[32][K] @ W[K][128] + b)
// Wave w owns output cols [16w,16w+16); computes 2 row-tiles of 16 moves.
// Wt is f16 [128][K] (transposed). In-place X==Y is safe (barrier-separated).
// ---------------------------------------------------------------------------
__device__ __forceinline__ void gemm_layer(
    const _Float16* __restrict__ Xs, int xstr, int ktot,
    const _Float16* __restrict__ Wt, const float* __restrict__ bias,
    bool do_relu, _Float16* __restrict__ Ys, int ystr, int lane, int wave) {

    const int l16 = lane & 15;
    const int hi  = lane >> 4;          // lane group 16..31
    v8f c0 = {}; v8f c1 = {};

    const _Float16* wrow = Wt + (size_t)(wave * 16 + l16) * ktot;
    const _Float16* x0   = Xs + (size_t)(l16)      * xstr;
    const _Float16* x1   = Xs + (size_t)(16 + l16) * xstr;
    const int boff = hi ? 16 : 0;       // unused: B frag is read at aoff, same K order as A
    const int aoff = hi ? 8  : 0;       // A frag: halves 0-7 -> K+0..7 / K+8..15

    #pragma unroll 4
    for (int kt = 0; kt < ktot / 32; ++kt) {
        const int kb = kt * 32;
        half8 bfl = *(const half8*)(wrow + kb + aoff);            // 16B global
        half8 bfh = *(const half8*)(wrow + kb + aoff + 16);       // 16B global
        half16 bf = __builtin_shufflevector(bfl, bfh,
                        0,1,2,3,4,5,6,7,8,9,10,11,12,13,14,15);

        half8 a0l = *(const half8*)(x0 + kb + aoff);              // 16B LDS
        half8 a0h = *(const half8*)(x0 + kb + aoff + 16);         // 16B LDS
        half16 a0 = __builtin_shufflevector(a0l, a0h,
                        0,1,2,3,4,5,6,7,8,9,10,11,12,13,14,15);
        c0 = __builtin_amdgcn_wmma_f32_16x16x32_f16(false, a0, false, bf,
                                                    (short)0, c0, false, false);

        half8 a1l = *(const half8*)(x1 + kb + aoff);
        half8 a1h = *(const half8*)(x1 + kb + aoff + 16);
        half16 a1 = __builtin_shufflevector(a1l, a1h,
                        0,1,2,3,4,5,6,7,8,9,10,11,12,13,14,15);
        c1 = __builtin_amdgcn_wmma_f32_16x16x32_f16(false, a1, false, bf,
                                                    (short)0, c1, false, false);
    }

    const float bv = bias[wave * 16 + l16];
    __syncthreads();                    // all reads of X done before writes
    #pragma unroll
    for (int r = 0; r < 8; ++r) {
        // C/D layout: VGPR r -> row r (lanes 0-15) / row r+8 (lanes 16-31)
        int row0 = r + (hi ? 8 : 0);
        float v0 = c0[r] + bv; if (do_relu) v0 = v0 > 0.f ? v0 : 0.f;
        Ys[(size_t)row0 * ystr + wave * 16 + l16] = (_Float16)v0;
        int row1 = 16 + row0;
        float v1 = c1[r] + bv; if (do_relu) v1 = v1 > 0.f ? v1 : 0.f;
        Ys[(size_t)row1 * ystr + wave * 16 + l16] = (_Float16)v1;
    }
    __syncthreads();
}

// ---------------------------------------------------------------------------
// Fused: gather E edges/move -> MLP1 (relu, linear) -> action net (3x relu)
// -> Wo dot -> logits.  Grid: (ceil(M/32), B).  256 threads = 8 waves.
// ---------------------------------------------------------------------------
template <int E>
__global__ __launch_bounds__(256) void vrp_fused_kernel(
    const float* __restrict__ e_emb, const int* __restrict__ idx,
    const _Float16* __restrict__ Wt1, const float* __restrict__ b1,
    const _Float16* __restrict__ Wt2, const float* __restrict__ b2,
    const _Float16* __restrict__ aWt1, const float* __restrict__ ab1,
    const _Float16* __restrict__ aWt2, const float* __restrict__ ab2,
    const _Float16* __restrict__ aWt3, const float* __restrict__ ab3,
    const _Float16* __restrict__ wo, const float* __restrict__ bo,
    float* __restrict__ out, int type_off) {

    constexpr int K1   = E * H_DIM;
    constexpr int FSTR = K1 + 8;     // padded feature row stride (halves)

    __shared__ _Float16 Xs[TM * FSTR];   // gathered features, f16
    __shared__ _Float16 Hs[TM * HSTR];   // hidden activations, f16 (in-place)

    const int chunk = blockIdx.x;
    const int b     = blockIdx.y;
    const int tid   = threadIdx.x;
    const int wave  = tid >> 5;
    const int lane  = tid & 31;

    // ---- gather: one (move,edge) row of 128 f32 per wave-iter, f32->f16 ----
    const int rows = TM * E;
    for (int r = wave; r < rows; r += 8) {
        int lm = r / E, e = r % E;
        int gm = chunk * TM + lm; if (gm > M_DIM - 1) gm = M_DIM - 1;  // tail clamp
        const int* ip = idx + (((size_t)b * M_DIM + gm) * E + e) * 2;
        int i = ip[0], j = ip[1];                       // uniform per wave
        const float4* src = (const float4*)(e_emb +
            (((size_t)b * N_DIM + i) * N_DIM + j) * H_DIM);
        float4 f = src[lane];                           // 16B coalesced
        half4_t h; h.x = (_Float16)f.x; h.y = (_Float16)f.y;
                   h.z = (_Float16)f.z; h.w = (_Float16)f.w;
        *(half4_t*)(&Xs[(size_t)lm * FSTR + e * H_DIM + lane * 4]) = h;
    }
    __syncthreads();

    // ---- MLP1: relu(X @ W1 + b1) @ W2 + b2 ----
    gemm_layer(Xs, FSTR, K1,    Wt1,  b1,  true,  Hs, HSTR, lane, wave);
    gemm_layer(Hs, HSTR, H_DIM, Wt2,  b2,  false, Hs, HSTR, lane, wave);
    // ---- action net: 3x relu(h @ W + b) ----
    gemm_layer(Hs, HSTR, H_DIM, aWt1, ab1, true,  Hs, HSTR, lane, wave);
    gemm_layer(Hs, HSTR, H_DIM, aWt2, ab2, true,  Hs, HSTR, lane, wave);
    gemm_layer(Hs, HSTR, H_DIM, aWt3, ab3, true,  Hs, HSTR, lane, wave);

    // ---- output head: logit = h . Wo + bo ----
    if (tid < TM) {
        int gm = chunk * TM + tid;
        if (gm < M_DIM) {
            float acc = bo[0];
            #pragma unroll 8
            for (int k = 0; k < H_DIM; ++k)
                acc += (float)Hs[(size_t)tid * HSTR + k] * (float)wo[k];
            out[(size_t)b * (3 * M_DIM) + type_off + gm] = acc;
        }
    }
}

// ---------------------------------------------------------------------------
extern "C" void kernel_launch(void* const* d_in, const int* in_sizes, int n_in,
                              void* d_out, int out_size, void* d_ws, size_t ws_size,
                              hipStream_t stream) {
    const float* e_emb     = (const float*)d_in[0];
    const int*   reloc_idx = (const int*)d_in[1];
    const int*   cross_idx = (const int*)d_in[2];
    const int*   twopt_idx = (const int*)d_in[3];
    const float* rW1 = (const float*)d_in[4];  const float* rb1 = (const float*)d_in[5];
    const float* rW2 = (const float*)d_in[6];  const float* rb2 = (const float*)d_in[7];
    const float* cW1 = (const float*)d_in[8];  const float* cb1 = (const float*)d_in[9];
    const float* cW2 = (const float*)d_in[10]; const float* cb2 = (const float*)d_in[11];
    const float* aW1 = (const float*)d_in[12]; const float* ab1 = (const float*)d_in[13];
    const float* aW2 = (const float*)d_in[14]; const float* ab2 = (const float*)d_in[15];
    const float* aW3 = (const float*)d_in[16]; const float* ab3 = (const float*)d_in[17];
    const float* aWo = (const float*)d_in[18]; const float* abo = (const float*)d_in[19];
    float* out = (float*)d_out;

    // workspace layout (f16 halves)
    _Float16* ws = (_Float16*)d_ws;
    _Float16* t_rW1 = ws;                       // [128][768]
    _Float16* t_rW2 = t_rW1 + 768 * 128;        // [128][128]
    _Float16* t_cW1 = t_rW2 + 128 * 128;        // [128][512]
    _Float16* t_cW2 = t_cW1 + 512 * 128;        // [128][128]
    _Float16* t_a1  = t_cW2 + 128 * 128;        // [128][128]
    _Float16* t_a2  = t_a1  + 128 * 128;
    _Float16* t_a3  = t_a2  + 128 * 128;
    _Float16* t_wo  = t_a3  + 128 * 128;        // [128]

    auto cvt = [&](const float* src, _Float16* dst, int K, int C) {
        int n = K * C;
        cvt_transpose_f16<<<(n + 255) / 256, 256, 0, stream>>>(src, dst, K, C);
    };
    cvt(rW1, t_rW1, 6 * H_DIM, H_DIM);
    cvt(rW2, t_rW2, H_DIM, H_DIM);
    cvt(cW1, t_cW1, 4 * H_DIM, H_DIM);
    cvt(cW2, t_cW2, H_DIM, H_DIM);
    cvt(aW1, t_a1,  H_DIM, H_DIM);
    cvt(aW2, t_a2,  H_DIM, H_DIM);
    cvt(aW3, t_a3,  H_DIM, H_DIM);
    cvt(aWo, t_wo,  H_DIM, 1);

    dim3 grid((M_DIM + TM - 1) / TM, B_DIM);   // (63, 64)
    vrp_fused_kernel<6><<<grid, 256, 0, stream>>>(
        e_emb, reloc_idx, t_rW1, rb1, t_rW2, rb2,
        t_a1, ab1, t_a2, ab2, t_a3, ab3, t_wo, abo, out, 0);
    vrp_fused_kernel<4><<<grid, 256, 0, stream>>>(
        e_emb, cross_idx, t_cW1, cb1, t_cW2, cb2,
        t_a1, ab1, t_a2, ab2, t_a3, ab3, t_wo, abo, out, M_DIM);
    vrp_fused_kernel<4><<<grid, 256, 0, stream>>>(
        e_emb, twopt_idx, t_cW1, cb1, t_cW2, cb2,
        t_a1, ab1, t_a2, ab2, t_a3, ab3, t_wo, abo, out, 2 * M_DIM);
}
